// MoE_62869731279220
// MI455X (gfx1250) — matrix operand order repaired and checked by reading, not compiled, not run on hardware
//
#include <hip/hip_runtime.h>

#define DMODEL   1024
#define NEXP     32
#define ESIZE    256
#define NHEADS   4
#define TOK      8192          // B * S
#define TILES    512           // worst case: one expert selected by all tokens

typedef __attribute__((ext_vector_type(16))) __bf16 v16bf;
typedef __attribute__((ext_vector_type(8)))  __bf16 v8bf;
typedef __attribute__((ext_vector_type(4)))  __bf16 v4bf;
typedef __attribute__((ext_vector_type(8)))  float  v8f;
typedef __attribute__((ext_vector_type(4)))  unsigned int v4u;
typedef __attribute__((ext_vector_type(8)))  int    v8i;
typedef __attribute__((ext_vector_type(4)))  int    v4i;

__device__ __forceinline__ v8f wmma_bf16(v16bf a, v16bf b, v8f c) {
    // D = A(16x32 bf16) * B(32x16 bf16) + C(16x16 f32)
    return __builtin_amdgcn_wmma_f32_16x16x32_bf16(false, a, false, b,
                                                   (short)0, c, false, false);
}

// ---------------------------------------------------------------------------
// Kernel 0: zero the output accumulator and the per-expert counters.
// ---------------------------------------------------------------------------
__global__ __launch_bounds__(256)
void zero_kernel(float4* __restrict__ y4, int n4, int* __restrict__ counts) {
    int i = blockIdx.x * blockDim.x + threadIdx.x;
    if (i < n4) y4[i] = make_float4(0.f, 0.f, 0.f, 0.f);
    if (i < NEXP) counts[i] = 0;
}

// ---------------------------------------------------------------------------
// Kernel 1: selection. One wave per 16 tokens.
// logits[16x32] = x_tile[16x1024] @ expert_sel^T via WMMA (2 N-tiles),
// sigmoid -> top-4 -> append (token, gate) to per-expert lists.
// ---------------------------------------------------------------------------
__global__ __launch_bounds__(32)
void sel_kernel(const float* __restrict__ x,
                const float* __restrict__ esel,
                int*   __restrict__ counts,
                int*   __restrict__ etok,
                float* __restrict__ egate) {
    __shared__ float sc[16 * 32];

    const int lane = threadIdx.x;
    const int hi = lane >> 4, lo = lane & 15;
    const int t0 = blockIdx.x * 16;

    const float* xrow = x + (size_t)(t0 + lo) * DMODEL;   // A row m = lo
    v8f acc0 = {}, acc1 = {};

#pragma unroll 1
    for (int ks = 0; ks < DMODEL / 32; ++ks) {
        v16bf a;
#pragma unroll
        for (int i = 0; i < 16; ++i) {
            int k = ks * 32 + (hi << 3) + (i & 7) + ((i >> 3) << 4);
            a[i] = (__bf16)xrow[k];
        }
        v16bf b0, b1;
#pragma unroll
        for (int i = 0; i < 16; ++i) {
            int k = ks * 32 + (hi << 3) + (i & 7) + ((i >> 3) << 4);             // B[k][n] = esel[n][k]
            b0[i] = (__bf16)esel[(size_t)lo * DMODEL + k];
            b1[i] = (__bf16)esel[(size_t)(16 + lo) * DMODEL + k];
        }
        acc0 = wmma_bf16(a, b0, acc0);
        acc1 = wmma_bf16(a, b1, acc1);
    }

#pragma unroll
    for (int r = 0; r < 8; ++r) {
        int m = r + (hi << 3);
        sc[m * 32 + lo]      = acc0[r];
        sc[m * 32 + 16 + lo] = acc1[r];
    }
    __syncthreads();

    if (lane < 16) {
        const int m = lane, tok = t0 + m;
        float bv[NHEADS] = {-1e30f, -1e30f, -1e30f, -1e30f};
        int   bi[NHEADS] = {0, 0, 0, 0};
#pragma unroll 1
        for (int e = 0; e < NEXP; ++e) {
            // gate-quality sigmoid: v_exp_f32 + v_rcp_f32 (no precise-div chain)
            float s = __builtin_amdgcn_rcpf(1.f + __expf(-sc[m * 32 + e]));
            if (s > bv[NHEADS - 1]) {                    // strict '>' keeps first index on ties
                bv[NHEADS - 1] = s; bi[NHEADS - 1] = e;
#pragma unroll
                for (int j = NHEADS - 1; j > 0; --j) {
                    if (bv[j] > bv[j - 1]) {
                        float tv = bv[j]; bv[j] = bv[j - 1]; bv[j - 1] = tv;
                        int   ti = bi[j]; bi[j] = bi[j - 1]; bi[j - 1] = ti;
                    }
                }
            }
        }
#pragma unroll
        for (int s = 0; s < NHEADS; ++s) {
            int e   = bi[s];
            int pos = atomicAdd(&counts[e], 1);
            etok [e * TOK + pos] = tok;
            egate[e * TOK + pos] = bv[s];
        }
    }
}

// ---------------------------------------------------------------------------
// Kernel 2: per-expert grouped GEMMs. Block = (tile of 16 gathered tokens,
// expert). 256 threads = 8 waves.
//   TDM gather-mode tensor_load_to_lds pulls the 16 token rows of x into LDS
//   (4 column chunks of 256 f32), a conflict-free pass converts to bf16.
//   GEMM1: H[16x256]  = gate * relu(Xg[16x1024] @ keys[e][1024x256])
//   GEMM2: Y[16x1024] = H @ values[e][256x1024]  -> atomicAdd scatter
// ---------------------------------------------------------------------------
__global__ __launch_bounds__(256)
void moe_kernel(const float* __restrict__ x,
                const float* __restrict__ keys,
                const float* __restrict__ values,
                const int*   __restrict__ counts,
                const int*   __restrict__ etok,
                const float* __restrict__ egate,
                float* __restrict__ y) {
    const int e    = blockIdx.y;
    const int tile = blockIdx.x;
    const int n    = counts[e];
    if (tile * 16 >= n) return;

    __shared__ float  stage[16][256];       // 16KB TDM gather destination
    __shared__ __bf16 xs[16][DMODEL + 8];   // +16B row pad: 4-bank rotation/row
    __shared__ __bf16 hs[16][ESIZE + 8];
    __shared__ int    stok[16];
    __shared__ float  sg[16];

    const int tid  = threadIdx.x;
    const int lane = tid & 31, wave = tid >> 5;
    const int hi   = lane >> 4, lo = lane & 15;

    if (tid < 16) {
        int idx = tile * 16 + tid;
        if (idx < n) { stok[tid] = etok[e * TOK + idx]; sg[tid] = egate[e * TOK + idx]; }
        else         { stok[tid] = etok[e * TOK];       sg[tid] = 0.f; }   // pad: gate 0
    }
    __syncthreads();

    // ---- TDM gather of 16 token rows (fp32), 4 chunks of 256 columns ------
    {
        // pack 16 row indices (tokens < 8192 fit u16) -> D# groups 2/3
        unsigned p[8];
#pragma unroll
        for (int j = 0; j < 8; ++j) {
            unsigned v = ((unsigned)stok[2 * j] & 0xffffu) |
                         ((unsigned)stok[2 * j + 1] << 16);
            p[j] = (unsigned)__builtin_amdgcn_readfirstlane((int)v);
        }
        v4i g2 = {(int)p[0], (int)p[1], (int)p[2], (int)p[3]};
        v4i g3 = {(int)p[4], (int)p[5], (int)p[6], (int)p[7]};

        // D# group1: data_size=4B, tensor 1024x8192, tile 256 x (16 indices),
        // dim0 stride = 1024 elements
        v8i g1 = {(int)0x00020000u,              // data_size=2 (4B)
                  (int)((DMODEL & 0xffff) << 16),// tensor_dim0 lo
                  (int)((TOK & 0xffff) << 16),   // tensor_dim0 hi=0 | tensor_dim1 lo
                  (int)(256u << 16),             // tensor_dim1 hi=0 | tile_dim0=256
                  (int)16,                       // tile_dim1 = #indices
                  (int)DMODEL,                   // tensor_dim0_stride lo
                  0, 0};

        v8i g4 = {0, 0, 0, 0, 0, 0, 0, 0};       // reserved/unused 5th group

        const unsigned lds = (unsigned)(size_t)&stage[0][0];

#pragma unroll 1
        for (int r = 0; r < 4; ++r) {
            if (tid < 32) {                      // wave 0 drives the TDM
                unsigned long long ga =
                    (unsigned long long)(size_t)x + (unsigned long long)r * 1024ull;
                v4u g0 = {0x80000001u,           // count=1 | gather_mode, 16-bit idx
                          lds,
                          (unsigned)(ga & 0xffffffffu),
                          (unsigned)((ga >> 32) & 0x01ffffffu) | 0x80000000u}; // type=2
                __builtin_amdgcn_tensor_load_to_lds(g0, g1, g2, g3, g4, 0);
                __builtin_amdgcn_s_wait_tensorcnt(0);
            }
            __syncthreads();
            // convert chunk to bf16 into padded xs (consecutive lanes ->
            // consecutive float4s -> conflict-free)
#pragma unroll
            for (int it = 0; it < 4; ++it) {
                int i4 = it * 256 + tid;         // float4 index in [16][256]
                int m  = i4 >> 6;
                int k  = (i4 & 63) * 4;
                float4 v = *(const float4*)&stage[m][k];
                v4bf w;
                w[0] = (__bf16)v.x; w[1] = (__bf16)v.y;
                w[2] = (__bf16)v.z; w[3] = (__bf16)v.w;
                *(v4bf*)&xs[m][r * 256 + k] = w;
            }
            __syncthreads();
        }
    }

    // ---------------- GEMM1: K=1024, N=256 (16 N-tiles, 2 per wave) ---------
    const float* Ke = keys + (size_t)e * DMODEL * ESIZE;
    v8f acc[2] = {{}, {}};
#pragma unroll 1
    for (int ks = 0; ks < DMODEL / 32; ++ks) {
        const int kb = ks * 32 + (hi << 3);
        v8bf a_lo = *(const v8bf*)&xs[lo][kb];
        v8bf a_hi = *(const v8bf*)&xs[lo][kb + 16];
        v16bf a = __builtin_shufflevector(a_lo, a_hi,
                    0, 1, 2, 3, 4, 5, 6, 7, 8, 9, 10, 11, 12, 13, 14, 15);
#pragma unroll
        for (int j = 0; j < 2; ++j) {
            const int col = (wave * 2 + j) * 16 + lo;
            v16bf b;
#pragma unroll
            for (int i = 0; i < 16; ++i) {
                int kk = ks * 32 + (hi << 3) + (i & 7) + ((i >> 3) << 4);
                b[i] = (__bf16)Ke[(size_t)kk * ESIZE + col];
            }
            acc[j] = wmma_bf16(a, b, acc[j]);
        }
    }
#pragma unroll
    for (int j = 0; j < 2; ++j)
#pragma unroll
        for (int r = 0; r < 8; ++r) {
            int m = r + (hi << 3);
            float h = acc[j][r];
            h = (h > 0.f ? h : 0.f) * sg[m];
            hs[m][(wave * 2 + j) * 16 + lo] = (__bf16)h;
        }
    __syncthreads();

    // ---------------- GEMM2: K=256, N=1024 (64 N-tiles, 8 per wave) ---------
    const float* Ve = values + (size_t)e * ESIZE * DMODEL;
#pragma unroll 1
    for (int j = 0; j < 8; ++j) {
        const int colb = (wave * 8 + j) * 16;
        v8f acc2 = {};
#pragma unroll 1
        for (int ks = 0; ks < ESIZE / 32; ++ks) {
            const int kb = ks * 32 + (hi << 3);
            v8bf a_lo = *(const v8bf*)&hs[lo][kb];
            v8bf a_hi = *(const v8bf*)&hs[lo][kb + 16];
            v16bf a = __builtin_shufflevector(a_lo, a_hi,
                        0, 1, 2, 3, 4, 5, 6, 7, 8, 9, 10, 11, 12, 13, 14, 15);
            v16bf b;
#pragma unroll
            for (int i = 0; i < 16; ++i) {
                int kk = ks * 32 + (hi << 3) + (i & 7) + ((i >> 3) << 4);
                b[i] = (__bf16)Ve[(size_t)kk * DMODEL + colb + lo];
            }
            acc2 = wmma_bf16(a, b, acc2);
        }
#pragma unroll
        for (int r = 0; r < 8; ++r) {
            int m = r + (hi << 3);
            atomicAdd(&y[(size_t)stok[m] * DMODEL + colb + lo], acc2[r]);
        }
    }
}

// ---------------------------------------------------------------------------
extern "C" void kernel_launch(void* const* d_in, const int* in_sizes, int n_in,
                              void* d_out, int out_size, void* d_ws, size_t ws_size,
                              hipStream_t stream) {
    const float* x      = (const float*)d_in[0];   // [4,2048,1024]
    const float* keys   = (const float*)d_in[1];   // [32,1024,256]
    const float* values = (const float*)d_in[2];   // [32,256,1024]
    const float* esel   = (const float*)d_in[3];   // [32,1024]
    float* y = (float*)d_out;                      // [4,2048,1024]

    // workspace: counts (32 int) | etok (32*8192 int) | egate (32*8192 f32)
    int*   counts = (int*)d_ws;
    int*   etok   = (int*)((char*)d_ws + 256);
    float* egate  = (float*)((char*)d_ws + 256 + (size_t)NEXP * TOK * sizeof(int));

    const int n4 = TOK * DMODEL / 4;
    zero_kernel<<<(n4 + 255) / 256, 256, 0, stream>>>((float4*)y, n4, counts);
    sel_kernel<<<TOK / 16, 32, 0, stream>>>(x, esel, counts, etok, egate);
    moe_kernel<<<dim3(TILES, NEXP), 256, 0, stream>>>(x, keys, values,
                                                      counts, etok, egate, y);
}
